// FPGAQuantizedLinear_9904194585389
// MI455X (gfx1250) — 8-bit matrix operand order repaired and checked by reading, compiled, not run on hardware
//
#include <hip/hip_runtime.h>


typedef int v8i __attribute__((ext_vector_type(8)));

#define ACT_SCALE   0.02f
#define EFF_SCALE   (0.02f * 0.01f)
#define NROWS       32768
#define KDIM        768
#define MDIM        768
#define KB          (KDIM / 64)
#define NB          (NROWS / 16)
#define MB          (MDIM / 16)
#define MG          (MDIM / 64)

__device__ __forceinline__ int a_koff(int vd, int g) {
    return ((vd >> 1) << 4) + ((vd & 1) << 2) + (g << 3);
}

__device__ __forceinline__ int b_koff(int vd, int g) {
    return ((vd >> 1) << 4) + ((vd & 1) << 2) + (g << 3);
}

__device__ __forceinline__ v8i load_frag(const signed char* __restrict__ p) {
    int4 lo = *(const int4*)(p);
    int4 hi = *(const int4*)(p + 512);
    v8i r;
    r[0] = lo.x; r[1] = lo.y; r[2] = lo.z; r[3] = lo.w;
    r[4] = hi.x; r[5] = hi.y; r[6] = hi.z; r[7] = hi.w;
    return r;
}

__global__ void quant_pack_a(const float* __restrict__ x,
                             signed char* __restrict__ apack) {
    int tid   = blockIdx.x * 256 + threadIdx.x;
    int inner = tid & 63;
    int blk   = tid >> 6;
    int c     = inner >> 5;
    int lane  = inner & 31;
    int g     = lane >> 4;
    int kb    = blk % KB;
    int nb    = blk / KB;
    int row   = nb * 16 + (lane & 15);

    const float* xr = x + (size_t)row * KDIM + kb * 64;

    unsigned int dw[4];
#pragma unroll
    for (int d = 0; d < 4; ++d) {
        int koff = a_koff(c * 4 + d, g);
        unsigned int w = 0;
#pragma unroll
        for (int j = 0; j < 4; ++j) {
            int q = __float2int_rn(xr[koff + j] * (1.0f / ACT_SCALE));
            q = q < -128 ? -128 : (q > 127 ? 127 : q);
            w |= ((unsigned int)(q & 0xff)) << (8 * j);
        }
        dw[d] = w;
    }
    int4 st = make_int4((int)dw[0], (int)dw[1], (int)dw[2], (int)dw[3]);
    *(int4*)(apack + (size_t)blk * 1024 + c * 512 + lane * 16) = st;
}

__global__ void pack_b(const signed char* __restrict__ w,
                       signed char* __restrict__ bpack) {
    int tid   = blockIdx.x * 256 + threadIdx.x;
    int inner = tid & 63;
    int blk   = tid >> 6;
    int c     = inner >> 5;
    int lane  = inner & 31;
    int g     = lane >> 4;
    int kb    = blk % KB;
    int mb    = blk / KB;
    int m     = mb * 16 + (lane & 15);

    unsigned int dw[4];
#pragma unroll
    for (int d = 0; d < 4; ++d) {
        int k = kb * 64 + b_koff(c * 4 + d, g);
        unsigned int wv = 0;
#pragma unroll
        for (int j = 0; j < 4; ++j) {
            signed char b = w[(size_t)(k + j) * MDIM + m];
            wv |= ((unsigned int)(unsigned char)b) << (8 * j);
        }
        dw[d] = wv;
    }
    int4 st = make_int4((int)dw[0], (int)dw[1], (int)dw[2], (int)dw[3]);
    *(int4*)(bpack + (size_t)blk * 1024 + c * 512 + lane * 16) = st;
}

__global__ void gemm_iu8(const signed char* __restrict__ apack,
                         const signed char* __restrict__ bpack,
                         const float* __restrict__ bias,
                         float* __restrict__ out) {
    int gwave = (blockIdx.x * 256 + threadIdx.x) >> 5;
    int lane  = threadIdx.x & 31;
    int npair = gwave / MG;
    int mg    = gwave % MG;
    int mt0   = mg * 4;

    const signed char* a0p = apack + (size_t)(2 * npair) * KB * 1024 + lane * 16;
    const signed char* a1p = a0p + (size_t)KB * 1024;
    const signed char* bp  = bpack + (size_t)mt0 * KB * 1024 + lane * 16;

    v8i acc0[4] = {};
    v8i acc1[4] = {};

    for (int kb = 0; kb < KB; ++kb) {
        v8i a0 = load_frag(a0p + kb * 1024);
        v8i a1 = load_frag(a1p + kb * 1024);
#pragma unroll
        for (int t = 0; t < 4; ++t) {
            v8i b = load_frag(bp + ((size_t)t * KB + kb) * 1024);
            acc0[t] = __builtin_amdgcn_wmma_i32_16x16x64_iu8(
                true, a0, true, b, acc0[t], false, false);
            acc1[t] = __builtin_amdgcn_wmma_i32_16x16x64_iu8(
                true, a1, true, b, acc1[t], false, false);
        }
    }

    int col_in = lane & 15;
    int rbase0 = 2 * npair * 16 + ((lane >> 4) << 3);
    int rbase1 = rbase0 + 16;
#pragma unroll
    for (int t = 0; t < 4; ++t) {
        int m = mt0 * 16 + t * 16 + col_in;
        float bv = bias[m];
#pragma unroll
        for (int i = 0; i < 8; ++i) {
            out[(size_t)(rbase0 + i) * MDIM + m] = (float)acc0[t][i] * EFF_SCALE + bv;
            out[(size_t)(rbase1 + i) * MDIM + m] = (float)acc1[t][i] * EFF_SCALE + bv;
        }
    }
}

extern "C" void kernel_launch(void* const* d_in, const int* in_sizes, int n_in,
                              void* d_out, int out_size, void* d_ws, size_t ws_size,
                              hipStream_t stream) {
    const float*       x    = (const float*)d_in[0];
    const signed char* wt   = (const signed char*)d_in[1];
    const float*       bias = (const float*)d_in[2];
    float*             out  = (float*)d_out;

    signed char* apack = (signed char*)d_ws;
    signed char* bpack = apack + (size_t)NROWS * KDIM;

    quant_pack_a<<<(NB * KB * 64) / 256, 256, 0, stream>>>(x, apack);

    pack_b<<<(MB * KB * 64) / 256, 256, 0, stream>>>(wt, bpack);

    gemm_iu8<<<((NB / 2) * MG) / 8, 256, 0, stream>>>(apack, bpack, bias, out);
}
